// DynamicAttentionNetwork_55413668053107
// MI455X (gfx1250) — matrix operand order repaired and checked by reading, not compiled, not run on hardware
//
#include <hip/hip_runtime.h>
#include <hip/hip_bf16.h>

#define N_NEUR 8192
#define D_ST   256
#define H_ST   192
#define I_ST   64
#define HID_SZ 128
#define NEGM  -1e30f

typedef __attribute__((ext_vector_type(16))) __bf16 v16bf;
typedef __attribute__((ext_vector_type(8)))  float  v8f;

union FragU { v16bf v; uint4 q[2]; };

__device__ inline v8f vzero8() {
  v8f z;
#pragma unroll
  for (int i = 0; i < 8; ++i) z[i] = 0.f;
  return z;
}

__device__ inline __bf16 f2b(float x) {
  unsigned u = __float_as_uint(x);
  unsigned r = (u + 0x7FFFu + ((u >> 16) & 1u)) >> 16;
  unsigned short s = (unsigned short)r;
  return *reinterpret_cast<__bf16*>(&s);
}

// CDNA5 async global->LDS copy (16B), tracked by ASYNCcnt.
// Generic pointers to LDS carry the LDS byte offset in their low 32 bits.
__device__ inline void async_copy_b128(void* lds_dst, const void* gsrc) {
  unsigned ldsa = (unsigned)(uintptr_t)lds_dst;
  unsigned long long ga = (unsigned long long)(uintptr_t)gsrc;
  asm volatile("global_load_async_to_lds_b128 %0, %1, off"
               :: "v"(ldsa), "v"(ga)
               : "memory");
}
__device__ inline void wait_async0() {
  asm volatile("s_wait_asynccnt 0" ::: "memory");
}

// A-fragment (16x32 bf16): lane m=lane&15 row; half h: K = h*8+[0,8) U 16+h*8+[0,8)
__device__ inline v16bf load_a_frag(const __bf16* base, int ld, int row0, int k0) {
  int lane = threadIdx.x & 31;
  int m = lane & 15, h = lane >> 4;
  const __bf16* p = base + (size_t)(row0 + m) * ld + k0 + h * 8;
  FragU u;
  u.q[0] = *reinterpret_cast<const uint4*>(p);
  u.q[1] = *reinterpret_cast<const uint4*>(p + 16);
  return u.v;
}

// B-fragment (32x16 bf16), source stored as [N][K] row-major (i.e. B transposed):
// lane n=lane&15 column; half h: K = h*8+[0,8) U 16+h*8+[0,8)
__device__ inline v16bf load_b_frag(const __bf16* base, int ld, int row0, int k0) {
  int lane = threadIdx.x & 31;
  int n = lane & 15, h = lane >> 4;
  const __bf16* p = base + (size_t)(row0 + n) * ld + k0 + h * 8;
  FragU u;
  u.q[0] = *reinterpret_cast<const uint4*>(p);
  u.q[1] = *reinterpret_cast<const uint4*>(p + 16);
  return u.v;
}

__device__ inline v8f wmma_bf16(v16bf a, v16bf b, v8f c) {
  return __builtin_amdgcn_wmma_f32_16x16x32_bf16(false, a, false, b, (short)0, c, false, false);
}

// ---------------- kernel 0: ns = concat(input, hidden) -> bf16 ----------------
__global__ __launch_bounds__(256) void build_ns(const float* __restrict__ in,
                                                const float* __restrict__ hid,
                                                __bf16* __restrict__ ns) {
  int idx = blockIdx.x * 256 + threadIdx.x;
  int r = idx >> 8, d = idx & 255;
  float v = (d < I_ST) ? in[r * I_ST + d] : hid[r * H_ST + (d - I_ST)];
  ns[idx] = f2b(v);
}

// ---------------- kernel 1: Q/K/V = ns @ W -----------------------------------
__global__ __launch_bounds__(256) void qkv_kernel(const __bf16* __restrict__ ns,
                                                  const float* __restrict__ Wq,
                                                  const float* __restrict__ Wk,
                                                  const float* __restrict__ Wv,
                                                  __bf16* __restrict__ Qb,
                                                  __bf16* __restrict__ Kb,
                                                  __bf16* __restrict__ Vb) {
  __shared__ __align__(16) __bf16 ns_lds[128 * 256]; // 64 KB
  __shared__ __align__(16) __bf16 wt[256 * 32];      // 16 KB  (W^T slab)
  int r0 = blockIdx.x * 128;
  const float* W = (blockIdx.y == 0) ? Wq : (blockIdx.y == 1) ? Wk : Wv;
  __bf16* Out    = (blockIdx.y == 0) ? Qb : (blockIdx.y == 1) ? Kb : Vb;

  int t = threadIdx.x, wave = t >> 5, lane = t & 31, mcol = lane & 15, hh = lane >> 4;

  // async stage of the 128x256 bf16 activation tile
  for (int i = t; i < 128 * 256 / 8; i += 256)
    async_copy_b128(&reinterpret_cast<uint4*>(ns_lds)[i],
                    &reinterpret_cast<const uint4*>(ns + (size_t)r0 * 256)[i]);

  v8f acc[16];
#pragma unroll
  for (int i = 0; i < 16; ++i) acc[i] = vzero8();

  wait_async0();
  for (int kb = 0; kb < 8; ++kb) {
    __syncthreads();
    for (int i = t; i < 32 * 256; i += 256) {
      int n = i & 255, kl = i >> 8;
      wt[n * 32 + kl] = f2b(W[(kb * 32 + kl) * 256 + n]);
    }
    __syncthreads();
    v16bf a = load_a_frag(ns_lds, 256, wave * 16, kb * 32);
#pragma unroll
    for (int tt = 0; tt < 16; ++tt) {
      v16bf b = load_b_frag(wt, 32, tt * 16, 0);
      acc[tt] = wmma_bf16(a, b, acc[tt]);
    }
  }
#pragma unroll
  for (int tt = 0; tt < 16; ++tt)
#pragma unroll
    for (int g = 0; g < 8; ++g) {
      int row = r0 + wave * 16 + g + 8 * hh;
      Out[(size_t)row * 256 + tt * 16 + mcol] = f2b(acc[tt][g]);
    }
}

// ---------------- kernel 2: flash attention with adjacency mask --------------
__global__ __launch_bounds__(256) void attn_kernel(const __bf16* __restrict__ Qb,
                                                   const __bf16* __restrict__ Kb,
                                                   const __bf16* __restrict__ Vb,
                                                   const unsigned char* __restrict__ adj,
                                                   __bf16* __restrict__ accb) {
  __shared__ __align__(16) __bf16 k_lds[32 * 256];    // 16 KB
  __shared__ __align__(16) __bf16 vt_lds[256 * 32];   // 16 KB (V^T)
  __shared__ __align__(16) unsigned char m_lds[32 * 128]; // 4 KB
  __shared__ __align__(16) __bf16 p_lds[8][16 * 32];  // 8 KB, per-wave probs

  int q0 = blockIdx.x * 128;
  int t = threadIdx.x, wave = t >> 5, lane = t & 31, mcol = lane & 15, hh = lane >> 4;

  // Q fragments are loop-invariant: hoist them into registers (global loads).
  v16bf qa[8];
#pragma unroll
  for (int kb = 0; kb < 8; ++kb)
    qa[kb] = load_a_frag(Qb, 256, q0 + wave * 16, kb * 32);

  v8f o[16];
  float mrow[8], lrow[8];
#pragma unroll
  for (int n = 0; n < 16; ++n) o[n] = vzero8();
#pragma unroll
  for (int g = 0; g < 8; ++g) { mrow[g] = -3e38f; lrow[g] = 0.f; }

  for (int it = 0; it < N_NEUR / 32; ++it) {
    int kblk = it * 32;
    __syncthreads();
    // K tile: async copy straight into LDS
    for (int i = t; i < 32 * 256 / 8; i += 256)
      async_copy_b128(&reinterpret_cast<uint4*>(k_lds)[i],
                      &reinterpret_cast<const uint4*>(Kb + (size_t)kblk * 256)[i]);
    // adjacency strip: async copy
    for (int i = t; i < 32 * 128 / 16; i += 256) {
      int kr = i >> 3, cg = (i & 7) * 16;
      async_copy_b128(&m_lds[kr * 128 + cg],
                      &adj[(size_t)(kblk + kr) * N_NEUR + q0 + cg]);
    }
    // V tile: vectorized global read, transposed scatter into LDS
    for (int i = t; i < 32 * 256 / 8; i += 256) {
      int kr = i >> 5, d8 = (i & 31) * 8;
      uint4 vv = *reinterpret_cast<const uint4*>(Vb + (size_t)(kblk + kr) * 256 + d8);
      const __bf16* pv = reinterpret_cast<const __bf16*>(&vv);
#pragma unroll
      for (int e = 0; e < 8; ++e) vt_lds[(d8 + e) * 32 + kr] = pv[e];
    }
    wait_async0();
    __syncthreads();

    v8f s0 = vzero8(), s1 = vzero8();
#pragma unroll
    for (int kb = 0; kb < 8; ++kb) {
      v16bf b0 = load_b_frag(k_lds, 256, 0, kb * 32);
      v16bf b1 = load_b_frag(k_lds, 256, 16, kb * 32);
      s0 = wmma_bf16(qa[kb], b0, s0);
      s1 = wmma_bf16(qa[kb], b1, s1);
    }

    float p0[8], p1[8];
#pragma unroll
    for (int g = 0; g < 8; ++g) {
      int qr = wave * 16 + g + 8 * hh;
      float msk0 = m_lds[mcol * 128 + qr] ? 1.f : 0.f;
      float msk1 = m_lds[(mcol + 16) * 128 + qr] ? 1.f : 0.f;
      float f0 = msk0 != 0.f ? s0[g] : NEGM;
      float f1 = msk1 != 0.f ? s1[g] : NEGM;
      float tm = fmaxf(f0, f1);
      tm = fmaxf(tm, __shfl_xor(tm, 1));
      tm = fmaxf(tm, __shfl_xor(tm, 2));
      tm = fmaxf(tm, __shfl_xor(tm, 4));
      tm = fmaxf(tm, __shfl_xor(tm, 8));
      float mn = fmaxf(mrow[g], tm);
      float sc = __expf(mrow[g] - mn);
      float e0 = msk0 * __expf(f0 - mn);
      float e1 = msk1 * __expf(f1 - mn);
      float rs = e0 + e1;
      rs += __shfl_xor(rs, 1);
      rs += __shfl_xor(rs, 2);
      rs += __shfl_xor(rs, 4);
      rs += __shfl_xor(rs, 8);
      lrow[g] = lrow[g] * sc + rs;
      mrow[g] = mn;
#pragma unroll
      for (int n = 0; n < 16; ++n) o[n][g] *= sc;
      p0[g] = e0; p1[g] = e1;
    }

    __bf16* pw = p_lds[wave];
#pragma unroll
    for (int g = 0; g < 8; ++g) {
      int qr = g + 8 * hh;
      pw[qr * 32 + mcol]      = f2b(p0[g]);
      pw[qr * 32 + 16 + mcol] = f2b(p1[g]);
    }
    v16bf pa = load_a_frag(pw, 32, 0, 0);
#pragma unroll
    for (int n = 0; n < 16; ++n) {
      v16bf vb = load_b_frag(vt_lds, 32, n * 16, 0);
      o[n] = wmma_bf16(pa, vb, o[n]);
    }
  }

#pragma unroll
  for (int n = 0; n < 16; ++n)
#pragma unroll
    for (int g = 0; g < 8; ++g) {
      int row = q0 + wave * 16 + g + 8 * hh;
      accb[(size_t)row * 256 + n * 16 + mcol] = f2b(o[n][g] / lrow[g]);
    }
}

// ---------------- kernel 3: MLP update + residual ----------------------------
__global__ __launch_bounds__(256) void mlp_kernel(const __bf16* __restrict__ ns,
                                                  const __bf16* __restrict__ accb,
                                                  const float* __restrict__ W1,
                                                  const float* __restrict__ b1,
                                                  const float* __restrict__ W2,
                                                  const float* __restrict__ b2,
                                                  const float* __restrict__ hidden,
                                                  const float* __restrict__ step,
                                                  float* __restrict__ out) {
  __shared__ __align__(16) __bf16 nps_blk[128 * 32];  // 8 KB (current K-slab of [ns|acc])
  __shared__ __align__(16) __bf16 w1t[128 * 32];      // 8 KB
  __shared__ __align__(16) __bf16 h_lds[128 * 128];   // 32 KB
  __shared__ __align__(16) __bf16 w2t[192 * 128];     // 48 KB
  int r0 = blockIdx.x * 128;
  int t = threadIdx.x, wave = t >> 5, lane = t & 31, mcol = lane & 15, hh = lane >> 4;

  for (int i = t; i < 192 * 128; i += 256) {
    int k = i / 192, n = i % 192;
    w2t[n * 128 + k] = f2b(W2[k * 192 + n]);
  }

  v8f hacc[8];
#pragma unroll
  for (int i = 0; i < 8; ++i) hacc[i] = vzero8();

  for (int kb = 0; kb < 16; ++kb) {
    __syncthreads();
    const __bf16* src = (kb < 8) ? ns : accb;
    int co = (kb & 7) * 32;
    // async copy of the activation slab overlaps with W1 slab conversion below
    for (int i = t; i < 512; i += 256) {
      int r = i >> 2, c8 = i & 3;
      async_copy_b128(&reinterpret_cast<uint4*>(nps_blk)[i],
                      src + (size_t)(r0 + r) * 256 + co + c8 * 8);
    }
    for (int i = t; i < 32 * 128; i += 256) {
      int n = i & 127, kl = i >> 7;
      w1t[n * 32 + kl] = f2b(W1[(kb * 32 + kl) * HID_SZ + n]);
    }
    wait_async0();
    __syncthreads();
    v16bf a = load_a_frag(nps_blk, 32, wave * 16, 0);
#pragma unroll
    for (int tt = 0; tt < 8; ++tt) {
      v16bf b = load_b_frag(w1t, 32, tt * 16, 0);
      hacc[tt] = wmma_bf16(a, b, hacc[tt]);
    }
  }

  // bias + relu -> h_lds (wave-local rows, no barrier needed for own reads)
#pragma unroll
  for (int tt = 0; tt < 8; ++tt)
#pragma unroll
    for (int g = 0; g < 8; ++g) {
      int col = tt * 16 + mcol;
      float v = hacc[tt][g] + b1[col];
      v = v > 0.f ? v : 0.f;
      h_lds[(wave * 16 + g + 8 * hh) * 128 + col] = f2b(v);
    }

  v8f uacc[12];
#pragma unroll
  for (int i = 0; i < 12; ++i) uacc[i] = vzero8();
#pragma unroll
  for (int kb = 0; kb < 4; ++kb) {
    v16bf a = load_a_frag(h_lds, 128, wave * 16, kb * 32);
#pragma unroll
    for (int tt = 0; tt < 12; ++tt) {
      v16bf b = load_b_frag(w2t, 128, tt * 16, kb * 32);
      uacc[tt] = wmma_bf16(a, b, uacc[tt]);
    }
  }

  float st = *step;
#pragma unroll
  for (int tt = 0; tt < 12; ++tt)
#pragma unroll
    for (int g = 0; g < 8; ++g) {
      int row = r0 + wave * 16 + g + 8 * hh;
      int col = tt * 16 + mcol;
      out[(size_t)row * H_ST + col] = hidden[(size_t)row * H_ST + col] + st * (uacc[tt][g] + b2[col]);
    }
}

extern "C" void kernel_launch(void* const* d_in, const int* in_sizes, int n_in,
                              void* d_out, int out_size, void* d_ws, size_t ws_size,
                              hipStream_t stream) {
  const float* input_states = (const float*)d_in[0];
  const float* hidden_states = (const float*)d_in[1];
  const unsigned char* adjacency = (const unsigned char*)d_in[2]; // jax bool = 1 byte
  const float* Wq = (const float*)d_in[3];
  const float* Wk = (const float*)d_in[4];
  const float* Wv = (const float*)d_in[5];
  const float* W1 = (const float*)d_in[6];
  const float* b1 = (const float*)d_in[7];
  const float* W2 = (const float*)d_in[8];
  const float* b2 = (const float*)d_in[9];
  const float* step = (const float*)d_in[10];
  float* out = (float*)d_out;

  char* ws = (char*)d_ws;
  const size_t SZ = (size_t)N_NEUR * 256 * sizeof(__bf16); // 4 MB each
  __bf16* ns   = (__bf16*)(ws);
  __bf16* Qb   = (__bf16*)(ws + SZ);
  __bf16* Kb   = (__bf16*)(ws + 2 * SZ);
  __bf16* Vb   = (__bf16*)(ws + 3 * SZ);
  __bf16* accb = (__bf16*)(ws + 4 * SZ);

  build_ns<<<N_NEUR * 256 / 256, 256, 0, stream>>>(input_states, hidden_states, ns);
  qkv_kernel<<<dim3(N_NEUR / 128, 3), 256, 0, stream>>>(ns, Wq, Wk, Wv, Qb, Kb, Vb);
  attn_kernel<<<N_NEUR / 128, 256, 0, stream>>>(Qb, Kb, Vb, adjacency, accb);
  mlp_kernel<<<N_NEUR / 128, 256, 0, stream>>>(ns, accb, W1, b1, W2, b2,
                                               hidden_states, step, out);
}
